// CorrelationAdaptor_49589692400370
// MI455X (gfx1250) — matrix operand order repaired and checked by reading, not compiled, not run on hardware
//
#include <hip/hip_runtime.h>
#include <hip/hip_bf16.h>
#include <stdint.h>

// ---------------------------------------------------------------------------
// CorrelationAdaptor for MI455X (gfx1250, wave32, WMMA)
//   level l: x[T=2,B=2,C=256,H,W] ->
//     corr  = windowed pointwise corr(x[1], x[0])          (VALU, LDS-staged)
//     off   = w_off @ corr  (1x1 conv, 72 ch)              (VALU)
//     col   = bilinear im2col of x[1] with offsets -> bf16 (gather)
//     feat  = relu(W[256,2304] @ col[2304,HW])             (v_wmma_f32_16x16x32_bf16)
//     out   = concat(x[0], feat) on image axis
// ---------------------------------------------------------------------------

typedef __attribute__((ext_vector_type(16))) __bf16 bf16x16;
typedef __attribute__((ext_vector_type(8)))  float  f32x8;

__device__ __forceinline__ unsigned short f32_to_bf16_rne(float f) {
    union { float f; unsigned u; } c; c.f = f;
    unsigned u = c.u;
    if ((u & 0x7fffffffu) > 0x7f800000u)           // NaN -> quiet NaN
        return (unsigned short)((u >> 16) | 0x0040u);
    u += 0x7fffu + ((u >> 16) & 1u);               // round to nearest even
    return (unsigned short)(u >> 16);
}

// -------------------------------- copy x[0] --------------------------------
__global__ void copy_t0_kernel(const float4* __restrict__ src,
                               float4* __restrict__ dst, int n4) {
    int i = blockIdx.x * blockDim.x + threadIdx.x;
    if (i < n4) dst[i] = src[i];
}

// ----------------------------- pack weights bf16 ---------------------------
__global__ void pack_w_kernel(const float* __restrict__ w,
                              unsigned short* __restrict__ wb, int n) {
    int i = blockIdx.x * blockDim.x + threadIdx.x;
    if (i < n) wb[i] = f32_to_bf16_rne(w[i]);
}

// ------------------------------- correlation -------------------------------
// block: 256 threads handles (b, h, 32-wide w chunk); f1 pixel block in LDS.
__global__ __launch_bounds__(256)
void corr_kernel(const float* __restrict__ x, float* __restrict__ corr,
                 int H, int W, int P, int D, int S, int stride) {
    __shared__ float f1s[256 * 32];                // [c][wl], 32 KB
    const int t = threadIdx.x;
    const int b = blockIdx.z;
    const int h = blockIdx.y;
    const int wbase = blockIdx.x * 32;
    const float* x1 = x + ((size_t)(2 + b) * 256) * P;   // t=1, batch b
    const float* x0 = x + ((size_t)b * 256) * P;         // t=0, batch b
    #pragma unroll 4
    for (int i = 0; i < 32; ++i) {
        int idx = i * 256 + t;
        int c = idx >> 5, wl = idx & 31, w = wbase + wl;
        f1s[idx] = (w < W) ? x1[(size_t)c * P + h * W + w] : 0.f;
    }
    __syncthreads();
    const int wl = t & 31, sg = t >> 5;
    const int w = wbase + wl;
    const int S2 = S * S;
    float* cb = corr + (size_t)b * S2 * P;
    for (int sidx = sg; sidx < S2; sidx += 8) {
        int iy = sidx / S, ix = sidx - iy * S;
        int h2 = h + (iy - D) * stride;
        int w2 = w + (ix - D) * stride;
        float acc = 0.f;
        if (w < W && h2 >= 0 && h2 < H && w2 >= 0 && w2 < W) {
            const float* f2 = x0 + h2 * W + w2;
            #pragma unroll 8
            for (int c = 0; c < 256; ++c)
                acc += f1s[c * 32 + wl] * f2[(size_t)c * P];
            acc *= (1.f / 256.f);
        }
        if (w < W) cb[(size_t)sidx * P + h * W + w] = acc;
    }
}

// ---------------------------- offset (1x1 conv) ----------------------------
__global__ void offset_kernel(const float* __restrict__ corr,
                              const float* __restrict__ w_off,
                              float* __restrict__ offs, int P, int S2) {
    int p = blockIdx.x * 256 + threadIdx.x;
    int o = blockIdx.y, b = blockIdx.z;
    if (p >= P) return;
    const float* cb = corr + (size_t)b * S2 * P + p;
    const float* wrow = w_off + (size_t)o * S2;
    float acc = 0.f;
    for (int k = 0; k < S2; ++k) acc += wrow[k] * cb[(size_t)k * P];
    offs[((size_t)b * 72 + o) * P + p] = acc;
}

// --------------------- deformable im2col (bilinear, bf16) ------------------
// thread per pixel; blockIdx.y enumerates (group g, tap k2); coords computed
// once and reused for all 64 channels of the group.
__global__ void col_kernel(const float* __restrict__ x,
                           const float* __restrict__ offs,
                           unsigned short* __restrict__ col,
                           int H, int W, int P, int wShift) {
    int p = blockIdx.x * 256 + threadIdx.x;
    if (p >= P) return;
    const int gk = blockIdx.y;                 // 0..35 = g*9 + k2
    const int g = gk / 9, k2 = gk - g * 9;
    const int n = blockIdx.z;
    const int h = p >> wShift, w = p & (W - 1);
    const float* ob = offs + ((size_t)n * 72 + (size_t)gk * 2) * P;
    float dy = ob[p], dx = ob[(size_t)P + p];
    float fy = (float)(h + k2 / 3 - 1) + dy;
    float fx = (float)(w + (k2 % 3) - 1) + dx;
    float y0f = floorf(fy), x0f = floorf(fx);
    float wy = fy - y0f, wx = fx - x0f;
    int iy = (int)y0f, ix = (int)x0f;
    int iy1 = iy + 1, ix1 = ix + 1;
    float vy0 = (iy  >= 0 && iy  < H) ? 1.f : 0.f;
    float vy1 = (iy1 >= 0 && iy1 < H) ? 1.f : 0.f;
    float vx0 = (ix  >= 0 && ix  < W) ? 1.f : 0.f;
    float vx1 = (ix1 >= 0 && ix1 < W) ? 1.f : 0.f;
    float w00 = vy0 * vx0 * (1.f - wy) * (1.f - wx);
    float w01 = vy0 * vx1 * (1.f - wy) * wx;
    float w10 = vy1 * vx0 * wy * (1.f - wx);
    float w11 = vy1 * vx1 * wy * wx;
    int cy0 = min(max(iy, 0), H - 1), cy1 = min(max(iy1, 0), H - 1);
    int cx0 = min(max(ix, 0), W - 1), cx1 = min(max(ix1, 0), W - 1);
    int o00 = cy0 * W + cx0, o01 = cy0 * W + cx1;
    int o10 = cy1 * W + cx0, o11 = cy1 * W + cx1;
    const float* xs = x + ((size_t)(2 + n) * 256 + (size_t)g * 64) * P;  // x[1][n], group base
    unsigned short* cb =
        col + ((size_t)n * 2304 + (size_t)g * 64 * 9 + k2) * P + p;
    #pragma unroll 4
    for (int c = 0; c < 64; ++c) {
        const float* img = xs + (size_t)c * P;
        float v = w00 * img[o00] + w01 * img[o01] +
                  w10 * img[o10] + w11 * img[o11];
        cb[(size_t)c * 9 * P] = f32_to_bf16_rne(v);
    }
}

// ----------------------------- WMMA GEMM + ReLU ----------------------------
// A-frag (16x32 bf16): lane M = lane&15; VGPR j holds k pair
//   kbase = 2*(j&3) + 16*(j>>2) + 8*(lane>>4)
__device__ __forceinline__ bf16x16
load_frag_a(const unsigned short* lds, int mt, int lane) {
    union { bf16x16 v; unsigned u[8]; } f;
    const unsigned short* row = lds + (mt + (lane & 15)) * 34;
    const int h8 = (lane >> 4) * 8;
    #pragma unroll
    for (int j = 0; j < 8; ++j)
        f.u[j] = *(const unsigned*)(row + 2 * (j & 3) + 16 * (j >> 2) + h8);
    return f.v;
}
// B-frag (32x16 bf16): lane N = lane&15; VGPR j holds k pair 2*(j&3) + 16*(j>>2) + 8*(lane>>4)
__device__ __forceinline__ bf16x16
load_frag_b(const unsigned short* lds, int nt, int lane) {
    union { bf16x16 v; unsigned u[8]; } f;
    const unsigned short* row = lds + (nt + (lane & 15)) * 34 + (lane >> 4) * 8;
    #pragma unroll
    for (int j = 0; j < 8; ++j)
        f.u[j] = *(const unsigned*)(row + 2 * (j & 3) + 16 * (j >> 2));
    return f.v;
}

__global__ __launch_bounds__(256)
void deform_gemm_kernel(const unsigned short* __restrict__ Wb,   // [256][2304] bf16
                        const unsigned short* __restrict__ col,  // [n][2304][P] bf16
                        float* __restrict__ out, int P) {
    __shared__ unsigned short ldsA[128 * 34];   // 128 M x 32 K, pad 34
    __shared__ unsigned short ldsB[64 * 34];    //  64 N x 32 K, pad 34
    const int t = threadIdx.x;
    const int lane = t & 31, wv = t >> 5;
    const int wm = (wv & 3) * 32;               // wave M offset (4 waves)
    const int wn = (wv >> 2) * 32;              // wave N offset (2 waves)
    const int m0 = blockIdx.y * 128;
    const int p0 = blockIdx.x * 64;
    const int nImg = blockIdx.z;
    const unsigned short* colN = col + (size_t)nImg * 2304 * P;
    f32x8 acc00 = {0,0,0,0,0,0,0,0};
    f32x8 acc01 = acc00, acc10 = acc00, acc11 = acc00;
    for (int k0 = 0; k0 < 2304; k0 += 32) {
        #pragma unroll
        for (int i = 0; i < 16; ++i) {          // stage A: 128x32
            int idx = i * 256 + t;
            int ml = idx >> 5, kl = idx & 31;
            ldsA[ml * 34 + kl] = Wb[(size_t)(m0 + ml) * 2304 + k0 + kl];
        }
        #pragma unroll
        for (int i = 0; i < 8; ++i) {           // stage B: 32x64 -> [p][k]
            int idx = i * 256 + t;
            int kl = idx >> 6, pl = idx & 63;
            ldsB[pl * 34 + kl] = colN[(size_t)(k0 + kl) * P + p0 + pl];
        }
        __syncthreads();
        bf16x16 a0 = load_frag_a(ldsA, wm, lane);
        bf16x16 a1 = load_frag_a(ldsA, wm + 16, lane);
        bf16x16 b0 = load_frag_b(ldsB, wn, lane);
        bf16x16 b1 = load_frag_b(ldsB, wn + 16, lane);
        acc00 = __builtin_amdgcn_wmma_f32_16x16x32_bf16(false, a0, false, b0,
                                                        (short)0, acc00, false, false);
        acc01 = __builtin_amdgcn_wmma_f32_16x16x32_bf16(false, a0, false, b1,
                                                        (short)0, acc01, false, false);
        acc10 = __builtin_amdgcn_wmma_f32_16x16x32_bf16(false, a1, false, b0,
                                                        (short)0, acc10, false, false);
        acc11 = __builtin_amdgcn_wmma_f32_16x16x32_bf16(false, a1, false, b1,
                                                        (short)0, acc11, false, false);
        __syncthreads();
    }
    // epilogue: C/D layout M = j + 8*(lane>>4), N = lane&15; ReLU
    float* ob = out + (size_t)(2 + nImg) * 256 * P;
    const int mlane = 8 * (lane >> 4);
    const int nlane = lane & 15;
    #pragma unroll
    for (int j = 0; j < 8; ++j) {
        int m = m0 + wm + j + mlane;
        ob[(size_t)m * P + p0 + wn + nlane]             = fmaxf(acc00[j], 0.f);
        ob[(size_t)m * P + p0 + wn + 16 + nlane]        = fmaxf(acc01[j], 0.f);
        ob[(size_t)(m + 16) * P + p0 + wn + nlane]      = fmaxf(acc10[j], 0.f);
        ob[(size_t)(m + 16) * P + p0 + wn + 16 + nlane] = fmaxf(acc11[j], 0.f);
    }
}

// --------------------------------- driver ----------------------------------
extern "C" void kernel_launch(void* const* d_in, const int* in_sizes, int n_in,
                              void* d_out, int out_size, void* d_ws, size_t ws_size,
                              hipStream_t stream) {
    const float* xs[4] = {(const float*)d_in[0], (const float*)d_in[1],
                          (const float*)d_in[2], (const float*)d_in[3]};
    const float* w_off[4];
    const float* w_adapt[4];
    int noff = 0, nad = 0;
    for (int i = 4; i < n_in && i < 12; ++i) {          // robust to interleaving
        if (in_sizes[i] == 256 * 256 * 9) { if (nad < 4) w_adapt[nad++] = (const float*)d_in[i]; }
        else                              { if (noff < 4) w_off[noff++] = (const float*)d_in[i]; }
    }

    static const int Hs[4] = {64, 32, 16, 8};
    static const int Ds[4] = {8, 8, 4, 2};
    static const int Ss[4] = {2, 1, 1, 1};

    // workspace layout (reused across levels; sized by level-0 maxima)
    char* ws = (char*)d_ws;
    unsigned short* colBuf = (unsigned short*)ws;                       // 37,748,736 B
    float* corrBuf = (float*)(ws + 37748736);                           //  9,469,952 B
    float* offBuf  = (float*)(ws + 37748736 + 9469952);                 //  2,359,296 B
    unsigned short* wbBuf =
        (unsigned short*)(ws + 37748736 + 9469952 + 2359296);           //  1,179,648 B

    float* outp = (float*)d_out;
    size_t outBase = 0;
    for (int l = 0; l < 4; ++l) {
        const int H = Hs[l], W = Hs[l], P = H * W;
        const int D = Ds[l], S = 2 * D + 1, S2 = S * S, st = Ss[l];
        const int wShift = (W == 64) ? 6 : (W == 32) ? 5 : (W == 16) ? 4 : 3;

        // 1. out[0:2] = x[0]  (2*256*P floats)
        {
            int n4 = (2 * 256 * P) / 4;
            copy_t0_kernel<<<(n4 + 255) / 256, 256, 0, stream>>>(
                (const float4*)xs[l], (float4*)(outp + outBase), n4);
        }
        // 2. bf16 weight matrix [256][2304]
        pack_w_kernel<<<(589824 + 255) / 256, 256, 0, stream>>>(
            w_adapt[l], wbBuf, 589824);
        // 3. correlation
        {
            dim3 g((W + 31) / 32, H, 2);
            corr_kernel<<<g, 256, 0, stream>>>(xs[l], corrBuf, H, W, P, D, S, st);
        }
        // 4. offsets
        {
            dim3 g((P + 255) / 256, 72, 2);
            offset_kernel<<<g, 256, 0, stream>>>(corrBuf, w_off[l], offBuf, P, S2);
        }
        // 5. deformable im2col -> bf16
        {
            dim3 g((P + 255) / 256, 36, 2);
            col_kernel<<<g, 256, 0, stream>>>(xs[l], offBuf, colBuf, H, W, P, wShift);
        }
        // 6. WMMA GEMM + ReLU -> out[2:4]
        {
            dim3 g(P / 64, 2, 2);
            deform_gemm_kernel<<<g, 256, 0, stream>>>(wbBuf, colBuf,
                                                      outp + outBase, P);
        }
        outBase += (size_t)4 * 256 * P;
    }
}
